// node_LSTM_40922448396350
// MI455X (gfx1250) — matrix operand order repaired and checked by reading, not compiled, not run on hardware
//
#include <hip/hip_runtime.h>
#include <hip/hip_bf16.h>
#include <math.h>

#define NROWS 65536
#define MDIM  512
#define HDIM  512
#define KCAT  1024   // M + H
#define OGATE 2048   // 4*H

typedef __attribute__((ext_vector_type(16))) __bf16 v16bf;
typedef __attribute__((ext_vector_type(8)))  float  v8f;

union V16BF {
    v16bf  v;
    float4 f[2];
};

__device__ __forceinline__ float sigmoidf_(float x) {
    return 1.0f / (1.0f + __expf(-x));
}

// ---------------------------------------------------------------------------
// Kernel 1: pack weights to bf16; W_cat = [m2h_w | h2h_w] (2048 x 1024),
// Wc2s (512x512), bias_cat = m2h_b + h2h_b.
// ---------------------------------------------------------------------------
__global__ __launch_bounds__(256)
void prep_w_kernel(const float* __restrict__ m2h_w, const float* __restrict__ h2h_w,
                   const float* __restrict__ c2s_w,
                   const float* __restrict__ m2h_b, const float* __restrict__ h2h_b,
                   __bf16* __restrict__ Wcat, __bf16* __restrict__ Wc2s,
                   float* __restrict__ bias_cat) {
    const int stride = gridDim.x * blockDim.x;
    int idx = blockIdx.x * blockDim.x + threadIdx.x;
    for (int i = idx; i < OGATE * KCAT; i += stride) {
        int o = i >> 10, k = i & (KCAT - 1);
        float v = (k < MDIM) ? m2h_w[o * MDIM + k] : h2h_w[o * HDIM + (k - MDIM)];
        Wcat[i] = (__bf16)v;
    }
    for (int i = idx; i < HDIM * HDIM; i += stride)
        Wc2s[i] = (__bf16)c2s_w[i];
    for (int i = idx; i < OGATE; i += stride)
        bias_cat[i] = m2h_b[i] + h2h_b[i];
}

// ---------------------------------------------------------------------------
// Kernel 2: per-row attention + bf16 packing. One wave32 per row.
// ---------------------------------------------------------------------------
__global__ __launch_bounds__(256)
void attn_prep_kernel(const float* __restrict__ msg, const float* __restrict__ hid,
                      const float* __restrict__ hid2, const float* __restrict__ cell,
                      const float* __restrict__ att_w, const float* __restrict__ att_b,
                      __bf16* __restrict__ Acat, __bf16* __restrict__ cellb) {
    const int row  = blockIdx.x * (blockDim.x >> 5) + (threadIdx.x >> 5);
    const int lane = threadIdx.x & 31;
    const size_t base = (size_t)row * HDIM;

    float lm = 0.f, l1 = 0.f, l2 = 0.f;
    for (int k = lane; k < HDIM; k += 32) {
        float m  = msg[base + k];
        float h1 = hid[base + k];
        float h2 = hid2[base + k];
        lm += m  * att_w[k];          // w_m
        l1 += h1 * att_w[MDIM + k];   // w_h
        l2 += h2 * att_w[MDIM + k];
        Acat[(size_t)row * KCAT + k] = (__bf16)m;
        cellb[base + k] = (__bf16)cell[base + k];
    }
    #pragma unroll
    for (int o = 16; o > 0; o >>= 1) {
        lm += __shfl_xor(lm, o, 32);
        l1 += __shfl_xor(l1, o, 32);
        l2 += __shfl_xor(l2, o, 32);
    }
    float b = att_b[0];
    float g1 = lm + b + l1, g2 = lm + b + l2;
    float mx = fmaxf(g1, g2);
    float e1 = __expf(g1 - mx), e2 = __expf(g2 - mx);
    float a1 = e1 / (e1 + e2), a2 = 1.0f - a1;

    for (int k = lane; k < HDIM; k += 32) {
        float hn = a1 * hid[base + k] + a2 * hid2[base + k];
        Acat[(size_t)row * KCAT + MDIM + k] = (__bf16)hn;
    }
}

// ---------------------------------------------------------------------------
// Kernel 3: cell path GEMM  (N x 512) @ (512 x 512)^T  via WMMA bf16.
// Each wave: 32 rows x 16 cols (2 accumulators sharing one B stream).
// Epilogue fuses tanh + delta-t update: cell_new = cell + tanh(acc+b)*(tdt-1)
// ---------------------------------------------------------------------------
__global__ __launch_bounds__(256)
void cell_gemm_kernel(const __bf16* __restrict__ cellb, const __bf16* __restrict__ Wc,
                      const float* __restrict__ c2s_b, const float* __restrict__ cell,
                      const float* __restrict__ tdt, float* __restrict__ cell_new) {
    const int wid  = blockIdx.x * (blockDim.x >> 5) + (threadIdx.x >> 5);
    const int lane = threadIdx.x & 31;
    const int colTiles = HDIM / 16;               // 32
    const int n0 = (wid / colTiles) * 32;         // two 16-row tiles
    const int c0 = (wid % colTiles) * 16;
    const int r    = lane & 15;
    const int half = lane >> 4;

    const __bf16* Ap0 = cellb + (size_t)(n0 + r) * HDIM + half * 16;
    const __bf16* Ap1 = cellb + (size_t)(n0 + 16 + r) * HDIM + half * 16;
    const __bf16* Bp  = Wc    + (size_t)(c0 + r) * HDIM + half * 16;

    v8f acc0 = {}, acc1 = {};
    for (int k0 = 0; k0 < HDIM; k0 += 32) {
        V16BF a0, a1;
        a0.f[0] = *(const float4*)(Ap0 + k0);
        a0.f[1] = *(const float4*)(Ap0 + k0 + 8);
        a1.f[0] = *(const float4*)(Ap1 + k0);
        a1.f[1] = *(const float4*)(Ap1 + k0 + 8);
        v16bf bmat = *(const v16bf*)(Bp + k0);
        acc0 = __builtin_amdgcn_wmma_f32_16x16x32_bf16(
                 false, a0.v, false, bmat, (short)0, acc0, false, false);
        acc1 = __builtin_amdgcn_wmma_f32_16x16x32_bf16(
                 false, a1.v, false, bmat, (short)0, acc1, false, false);
    }

    const int cc = c0 + (lane & 15);
    const float bcol = c2s_b[cc];
    #pragma unroll
    for (int v = 0; v < 8; ++v) {
        int m0 = n0 + v + half * 8;
        size_t idx0 = (size_t)m0 * HDIM + cc;
        float cs0 = tanhf(acc0[v] + bcol);
        cell_new[idx0] = cell[idx0] + cs0 * (tdt[idx0] - 1.0f);

        size_t idx1 = idx0 + (size_t)16 * HDIM;
        float cs1 = tanhf(acc1[v] + bcol);
        cell_new[idx1] = cell[idx1] + cs1 * (tdt[idx1] - 1.0f);
    }
}

// ---------------------------------------------------------------------------
// Kernel 4: gates GEMM  (N x 1024) @ (2048 x 1024)^T  via WMMA bf16.
// Each wave: 32 rows x 16 h-cols, 8 accumulators = {2 row tiles} x {i,f,g,o}
// gates at the same h (B tiles reused across both row tiles -> ~21 FLOP/B).
// Block = 8 waves covering 64 rows x 64 h. Full LSTM epilogue fused.
// ---------------------------------------------------------------------------
__global__ __launch_bounds__(256)
void gates_gemm_kernel(const __bf16* __restrict__ Acat, const __bf16* __restrict__ Wcat,
                       const float* __restrict__ bias, const float* __restrict__ cell_new,
                       float* __restrict__ hidden_out, float* __restrict__ cell_out) {
    const int w    = threadIdx.x >> 5;
    const int lane = threadIdx.x & 31;
    const int hb   = blockIdx.x & 7;              // 512/64 = 8 h-blocks
    const int rb   = blockIdx.x >> 3;
    const int n0   = rb * 64 + (w >> 2) * 32;     // two 16-row tiles
    const int h0   = hb * 64 + (w & 3) * 16;
    const int r    = lane & 15;
    const int half = lane >> 4;

    const __bf16* Ap0 = Acat + (size_t)(n0 + r) * KCAT + half * 16;
    const __bf16* Ap1 = Acat + (size_t)(n0 + 16 + r) * KCAT + half * 16;
    const __bf16* Bp  = Wcat + (size_t)(h0 + r) * KCAT + half * 16;  // gate i rows
    const size_t gstep = (size_t)HDIM * KCAT;                        // +512 out rows

    v8f a0i = {}, a0f = {}, a0g = {}, a0o = {};
    v8f a1i = {}, a1f = {}, a1g = {}, a1o = {};
    for (int k0 = 0; k0 < KCAT; k0 += 32) {
        V16BF a0, a1;
        a0.f[0] = *(const float4*)(Ap0 + k0);
        a0.f[1] = *(const float4*)(Ap0 + k0 + 8);
        a1.f[0] = *(const float4*)(Ap1 + k0);
        a1.f[1] = *(const float4*)(Ap1 + k0 + 8);
        v16bf b0 = *(const v16bf*)(Bp + k0);
        v16bf b1 = *(const v16bf*)(Bp + gstep + k0);
        v16bf b2 = *(const v16bf*)(Bp + 2 * gstep + k0);
        v16bf b3 = *(const v16bf*)(Bp + 3 * gstep + k0);
        a0i = __builtin_amdgcn_wmma_f32_16x16x32_bf16(false, a0.v, false, b0, (short)0, a0i, false, false);
        a1i = __builtin_amdgcn_wmma_f32_16x16x32_bf16(false, a1.v, false, b0, (short)0, a1i, false, false);
        a0f = __builtin_amdgcn_wmma_f32_16x16x32_bf16(false, a0.v, false, b1, (short)0, a0f, false, false);
        a1f = __builtin_amdgcn_wmma_f32_16x16x32_bf16(false, a1.v, false, b1, (short)0, a1f, false, false);
        a0g = __builtin_amdgcn_wmma_f32_16x16x32_bf16(false, a0.v, false, b2, (short)0, a0g, false, false);
        a1g = __builtin_amdgcn_wmma_f32_16x16x32_bf16(false, a1.v, false, b2, (short)0, a1g, false, false);
        a0o = __builtin_amdgcn_wmma_f32_16x16x32_bf16(false, a0.v, false, b3, (short)0, a0o, false, false);
        a1o = __builtin_amdgcn_wmma_f32_16x16x32_bf16(false, a1.v, false, b3, (short)0, a1o, false, false);
    }

    const int hc = h0 + (lane & 15);
    const float bi = bias[hc];
    const float bf = bias[HDIM + hc];
    const float bg = bias[2 * HDIM + hc];
    const float bo = bias[3 * HDIM + hc];

    #pragma unroll
    for (int v = 0; v < 8; ++v) {
        // row tile 0
        {
            int m = n0 + v + half * 8;
            size_t idx = (size_t)m * HDIM + hc;
            float iv = sigmoidf_(a0i[v] + bi);
            float fv = sigmoidf_(a0f[v] + bf);
            float gv = tanhf(a0g[v] + bg);
            float ov = sigmoidf_(a0o[v] + bo);
            float cn = cell_new[idx];
            float co = fv * cn + iv * gv;
            hidden_out[idx] = ov * tanhf(co);
            cell_out[idx]   = co;
        }
        // row tile 1 (+16 rows)
        {
            int m = n0 + 16 + v + half * 8;
            size_t idx = (size_t)m * HDIM + hc;
            float iv = sigmoidf_(a1i[v] + bi);
            float fv = sigmoidf_(a1f[v] + bf);
            float gv = tanhf(a1g[v] + bg);
            float ov = sigmoidf_(a1o[v] + bo);
            float cn = cell_new[idx];
            float co = fv * cn + iv * gv;
            hidden_out[idx] = ov * tanhf(co);
            cell_out[idx]   = co;
        }
    }
}

// ---------------------------------------------------------------------------
// Launch
// ---------------------------------------------------------------------------
extern "C" void kernel_launch(void* const* d_in, const int* in_sizes, int n_in,
                              void* d_out, int out_size, void* d_ws, size_t ws_size,
                              hipStream_t stream) {
    const float* message = (const float*)d_in[0];
    const float* hidden  = (const float*)d_in[1];
    const float* cell    = (const float*)d_in[2];
    const float* tdt     = (const float*)d_in[3];
    const float* hidden2 = (const float*)d_in[4];
    const float* m2h_w   = (const float*)d_in[5];
    const float* m2h_b   = (const float*)d_in[6];
    const float* h2h_w   = (const float*)d_in[7];
    const float* h2h_b   = (const float*)d_in[8];
    const float* c2s_w   = (const float*)d_in[9];
    const float* c2s_b   = (const float*)d_in[10];
    const float* att_w   = (const float*)d_in[11];
    const float* att_b   = (const float*)d_in[12];

    float* out        = (float*)d_out;
    float* hidden_out = out;                             // first return value
    float* cell_out   = out + (size_t)NROWS * HDIM;      // second return value

    // workspace carve-up (offsets 256B aligned)
    char* ws = (char*)d_ws;
    __bf16* Acat   = (__bf16*)(ws);                                   // 134,217,728 B
    __bf16* cellb  = (__bf16*)(ws + 134217728ull);                    //  67,108,864 B
    float*  cnew   = (float*) (ws + 134217728ull + 67108864ull);      // 134,217,728 B
    __bf16* Wcat   = (__bf16*)(ws + 335544320ull);                    //   4,194,304 B
    __bf16* Wc2s   = (__bf16*)(ws + 339738624ull);                    //     524,288 B
    float*  biasc  = (float*) (ws + 340262912ull);                    //       8,192 B

    // 1. weight packing (grid-stride)
    prep_w_kernel<<<4096, 256, 0, stream>>>(m2h_w, h2h_w, c2s_w, m2h_b, h2h_b,
                                            Wcat, Wc2s, biasc);

    // 2. attention + bf16 packing: 1 wave per row, 8 waves per block
    attn_prep_kernel<<<NROWS / 8, 256, 0, stream>>>(message, hidden, hidden2, cell,
                                                    att_w, att_b, Acat, cellb);

    // 3. cell GEMM: (N/32)*(512/16) = 65536 wave-tiles, 8 waves per block
    cell_gemm_kernel<<<(NROWS / 32) * (HDIM / 16) / 8, 256, 0, stream>>>(
        cellb, Wc2s, c2s_b, cell, tdt, cnew);

    // 4. gates GEMM + fused LSTM epilogue: (N/64)*(512/64) blocks
    gates_gemm_kernel<<<(NROWS / 64) * (HDIM / 64), 256, 0, stream>>>(
        Acat, Wcat, biasc, cnew, hidden_out, cell_out);
}
